// VRWKV_SpatialMix_75977971466439
// MI455X (gfx1250) — matrix operand order repaired and checked by reading, not compiled, not run on hardware
//
#include <hip/hip_runtime.h>

typedef __attribute__((ext_vector_type(16))) __bf16 v16bf;
typedef __attribute__((ext_vector_type(8)))  __bf16 v8bf;
typedef __attribute__((ext_vector_type(8)))  float  v8f;
typedef int vi4 __attribute__((vector_size(16)));   // 16B unit for async-to-LDS

namespace {
constexpr int kB = 8;
constexpr int kT = 4096;          // 64*64
constexpr int kC = 768;
constexpr long kBT = (long)kB * kT;   // 32768 rows
}

// SHIFTS table (dy, dx) for the 24 channel groups of 32.
__constant__ int c_dy[24] = {0,0,1,-1,1,-1,1,-1,0,0,2,-2,2,-2,2,-2,2,1,2,1,-2,-1,-2,-1};
__constant__ int c_dx[24] = {1,-1,0,0,1,-1,-1,1,2,-2,0,0,2,-2,-2,2,1,2,-1,-2,1,2,-1,-2};

// zigzag (top-left, horizontal): spatial position of sequence index t
__device__ __forceinline__ int zig(int t) {
  int r = t >> 6, j = t & 63;
  int col = (r & 1) ? (63 - j) : j;
  return (r << 6) | col;
}

// ---------------------------------------------------------------------------
// global(16B) -> LDS(16B) copy: async-to-LDS (ASYNCcnt path) if the toolchain
// exposes it, otherwise plain load + ds_store_b128.
// ---------------------------------------------------------------------------
__device__ __forceinline__ void copy16_g2l(__bf16* lds_dst, const __bf16* gsrc) {
#if __has_builtin(__builtin_amdgcn_global_load_async_to_lds_b128)
  __builtin_amdgcn_global_load_async_to_lds_b128(
      (__attribute__((address_space(1))) vi4*)gsrc,
      (__attribute__((address_space(3))) vi4*)lds_dst, 0, 0);
#else
  *reinterpret_cast<v8bf*>(lds_dst) = *reinterpret_cast<const v8bf*>(gsrc);
#endif
}

__device__ __forceinline__ void wait_async_done() {
#if __has_builtin(__builtin_amdgcn_global_load_async_to_lds_b128)
#if __has_builtin(__builtin_amdgcn_s_wait_asynccnt)
  __builtin_amdgcn_s_wait_asynccnt(0);
#else
  asm volatile("s_wait_asynccnt 0" ::: "memory");
#endif
#endif
}

// ---------------------------------------------------------------------------
// Kernel 1: fused mul_shift + zigzag gather, fp32 -> bf16, 4 channels/thread
// ---------------------------------------------------------------------------
__global__ void prep_xs(const float* __restrict__ x, __bf16* __restrict__ xs) {
  long g = (long)blockIdx.x * blockDim.x + threadIdx.x;   // [0, B*T*C/4)
  int  c4 = (int)(g % (kC / 4));
  long bs = g / (kC / 4);
  int  s  = (int)(bs % kT);
  int  b  = (int)(bs / kT);
  int  c0 = c4 * 4;
  int  grp = c0 >> 5;                    // 32 channels per shift group
  int  r = s >> 6, j = s & 63;
  int  xc = (r & 1) ? (63 - j) : j;      // spatial coords of zigzag position s
  int  sy = r  - c_dy[grp];
  int  sx = xc - c_dx[grp];
  float4 val = make_float4(0.f, 0.f, 0.f, 0.f);
  if ((unsigned)sy < 64u && (unsigned)sx < 64u)
    val = *reinterpret_cast<const float4*>(
        x + ((long)b * kT + sy * 64 + sx) * kC + c0);
  union { __bf16 h[4]; uint2 u2; } u;
  u.h[0] = (__bf16)val.x; u.h[1] = (__bf16)val.y;
  u.h[2] = (__bf16)val.z; u.h[3] = (__bf16)val.w;
  *reinterpret_cast<uint2*>(xs + ((long)b * kT + s) * kC + c0) = u.u2;
}

// ---------------------------------------------------------------------------
// Kernel 2: convert the 4 weight matrices to bf16 (kept row-major (n, c))
// ---------------------------------------------------------------------------
__global__ void conv_w(const float* __restrict__ kw, const float* __restrict__ vw,
                       const float* __restrict__ rw, const float* __restrict__ ow,
                       __bf16* __restrict__ wb) {
  long g = (long)blockIdx.x * blockDim.x + threadIdx.x;   // [0, 4*768*768/4)
  constexpr long kQ = (long)kC * kC / 4;
  int  mat = (int)(g / kQ);
  long rem = g % kQ;
  const float* src = (mat == 0) ? kw : (mat == 1) ? vw : (mat == 2) ? rw : ow;
  float4 val = *reinterpret_cast<const float4*>(src + rem * 4);
  union { __bf16 h[4]; uint2 u2; } u;
  u.h[0] = (__bf16)val.x; u.h[1] = (__bf16)val.y;
  u.h[2] = (__bf16)val.z; u.h[3] = (__bf16)val.w;
  *reinterpret_cast<uint2*>(wb + (long)mat * kC * kC + rem * 4) = u.u2;
}

// ---------------------------------------------------------------------------
// A-fragment loader (global): A 16x32 bf16 per CDNA5 ISA §7.12.2:
// lane holds row M=lane&15; lanes<16: K 0-7 & 16-23, lanes>=16: K 8-15 & 24-31.
// ---------------------------------------------------------------------------
__device__ __forceinline__ v16bf load_a_frag(const __bf16* __restrict__ row,
                                             int k0, int hi) {
  v8bf a0 = *reinterpret_cast<const v8bf*>(row + k0 + hi * 8);
  v8bf a1 = *reinterpret_cast<const v8bf*>(row + k0 + 16 + hi * 8);
  return __builtin_shufflevector(a0, a1, 0,1,2,3,4,5,6,7,8,9,10,11,12,13,14,15);
}

// ---------------------------------------------------------------------------
// WMMA GEMM over NW weight matrices sharing one A operand.
// Block = 8 waves (256 thr); wave tile = 16 rows x 64 cols per weight.
// The whole B slab for the block (NW x 64 rows x 768 K, bf16) is staged ONCE
// into LDS, pre-swizzled into WMMA-fragment order so that each ds_load_b128
// is a perfectly sequential 512B wave access (conflict-free):
//   unit layout: ((ks*NW*4 + wsel*4 + nt)*64 + jh*32 + lane) * 16B
//   where ks = K-step (32 K), jh selects the 8-element half of a lane's run,
//   lane = (n&15) + 16*(K-half) — exactly the B 32x16 operand layout.
// ---------------------------------------------------------------------------
template <int NW>
__global__ __launch_bounds__(256) void gemm_wmma(
    const __bf16* __restrict__ A, const __bf16* __restrict__ Wb,
    float* __restrict__ out0, float* __restrict__ out1,
    __bf16* __restrict__ out2) {
  extern __shared__ __bf16 smem[];          // NW * 96KB
  const int lane = threadIdx.x & 31;
  const int wave = threadIdx.x >> 5;
  const int hi   = lane >> 4;
  const int m0   = blockIdx.x * 128 + wave * 16;
  const int n0   = blockIdx.y * 64;

  // ---- stage B slab into LDS (async if available) ----
  for (int i = threadIdx.x; i < NW * 6144; i += 256) {
    int wsel    = i / 6144;
    int rem     = i % 6144;
    int n_local = rem / 96;                 // 0..63
    int k       = (rem % 96) * 8;           // 8-element chunk along K
    int ks      = k >> 5;
    int khalf   = (k >> 3) & 1;
    int jh      = (k >> 4) & 1;
    int lane_t  = (n_local & 15) | (khalf << 4);
    int f       = wsel * 4 + (n_local >> 4);
    int unit    = (ks * (NW * 4) + f) * 64 + jh * 32 + lane_t;
    copy16_g2l(smem + unit * 8,
               Wb + (long)wsel * kC * kC + (long)(n0 + n_local) * kC + k);
  }
  wait_async_done();
  __syncthreads();

  // ---- main WMMA loop: A from global (L2-resident), B from LDS ----
  const __bf16* arow = A + (long)(m0 + (lane & 15)) * kC;
  const v8f z = {0.f,0.f,0.f,0.f,0.f,0.f,0.f,0.f};
  v8f acc[NW][4];
  #pragma unroll
  for (int i = 0; i < NW; ++i)
    #pragma unroll
    for (int t = 0; t < 4; ++t) acc[i][t] = z;

  #pragma unroll 1
  for (int ks = 0; ks < kC / 32; ++ks) {
    v16bf a = load_a_frag(arow, ks * 32, hi);
    #pragma unroll
    for (int wsel = 0; wsel < NW; ++wsel) {
      #pragma unroll
      for (int nt = 0; nt < 4; ++nt) {
        const __bf16* fb = smem + ((ks * (NW * 4) + wsel * 4 + nt) * 64) * 8;
        v8bf b0 = *reinterpret_cast<const v8bf*>(fb + lane * 8);
        v8bf b1 = *reinterpret_cast<const v8bf*>(fb + (32 + lane) * 8);
        v16bf bfrag = __builtin_shufflevector(
            b0, b1, 0,1,2,3,4,5,6,7,8,9,10,11,12,13,14,15);
        acc[wsel][nt] = __builtin_amdgcn_wmma_f32_16x16x32_bf16(
            false, a, false, bfrag, (short)0, acc[wsel][nt], false, false);
      }
    }
  }

  // ---- epilogue: C/D row = m0 + hi*8 + r, col = n0 + nt*16 + (lane&15) ----
  const int outm = m0 + hi * 8;
  #pragma unroll
  for (int nt = 0; nt < 4; ++nt) {
    int n = n0 + nt * 16 + (lane & 15);
    #pragma unroll
    for (int r = 0; r < 8; ++r) {
      long off = (long)(outm + r) * kC + n;
      if constexpr (NW == 3) {
        out0[off] = acc[0][nt][r];
        out1[off] = acc[1][nt][r];
        float s = acc[2][nt][r];
        out2[off] = (__bf16)(1.0f / (1.0f + __expf(-s)));  // fused sigmoid
      } else {
        out0[off] = acc[0][nt][r];
      }
    }
  }
}

// ---------------------------------------------------------------------------
// WKV recurrence, one thread per (b, c), in-place over v.
// Scan 2 gathers/writes at zig(t): y2[t] lands at final position order[t].
// ---------------------------------------------------------------------------
__global__ void wkv_scan(const float* __restrict__ k, float* __restrict__ v,
                         const float* __restrict__ decay,
                         const float* __restrict__ first,
                         int sel, int useOrder) {
  int g = blockIdx.x * blockDim.x + threadIdx.x;   // [0, B*C)
  int c = g % kC;
  int b = g / kC;
  const float w = decay[sel * kC + c] * (1.0f / kT);
  const float u = first[sel * kC + c] * (1.0f / kT);
  const float* kb = k + (long)b * kT * kC + c;
  float*       vb = v + (long)b * kT * kC + c;
  float p = 0.f, q = 0.f, o = -1e38f;
  #pragma unroll 1
  for (int t = 0; t < kT; ++t) {
    int  s   = useOrder ? zig(t) : t;
    long off = (long)s * kC;
    float kt = kb[off];
    float vt = vb[off];
    float no  = fmaxf(o, u + kt);
    float A   = __expf(o - no);
    float Bc  = __expf(u + kt - no);
    float y   = (A * p + Bc * vt) / (A * q + Bc);
    float no2 = fmaxf(w + o, kt);
    float A2  = __expf(w + o - no2);
    float B2  = __expf(kt - no2);
    p = A2 * p + B2 * vt;
    q = A2 * q + B2;
    o = no2;
    vb[off] = y;                                   // read-before-write: safe
  }
}

// ---------------------------------------------------------------------------
// rv = bf16(sr * v)
// ---------------------------------------------------------------------------
__global__ void mul_sr(const __bf16* __restrict__ sr, const float* __restrict__ v,
                       __bf16* __restrict__ rv) {
  long g = ((long)blockIdx.x * blockDim.x + threadIdx.x) * 4;  // element index
  union { __bf16 h[4]; uint2 u2; } si;
  si.u2 = *reinterpret_cast<const uint2*>(sr + g);
  float4 vv = *reinterpret_cast<const float4*>(v + g);
  union { __bf16 h[4]; uint2 u2; } out;
  out.h[0] = (__bf16)((float)si.h[0] * vv.x);
  out.h[1] = (__bf16)((float)si.h[1] * vv.y);
  out.h[2] = (__bf16)((float)si.h[2] * vv.z);
  out.h[3] = (__bf16)((float)si.h[3] * vv.w);
  *reinterpret_cast<uint2*>(rv + g) = out.u2;
}

// ---------------------------------------------------------------------------
extern "C" void kernel_launch(void* const* d_in, const int* in_sizes, int n_in,
                              void* d_out, int out_size, void* d_ws, size_t ws_size,
                              hipStream_t stream) {
  (void)in_sizes; (void)n_in; (void)out_size; (void)ws_size;

  const float* x      = (const float*)d_in[0];
  const float* key_w  = (const float*)d_in[1];
  const float* val_w  = (const float*)d_in[2];
  const float* rec_w  = (const float*)d_in[3];
  const float* out_w  = (const float*)d_in[4];
  const float* decay  = (const float*)d_in[5];   // (2, C)
  const float* first  = (const float*)d_in[6];   // (2, C)

  char* ws = (char*)d_ws;
  const long nElem   = kBT * kC;                 // 25,165,824
  __bf16* xs   = (__bf16*)ws;                    // bf16 activations / reused as rv
  __bf16* wb   = (__bf16*)(ws + nElem * 2);      // 4 bf16 weight matrices
  float*  kf   = (float*)(ws + nElem * 2 + 4L * kC * kC * 2);
  float*  vf   = kf + nElem;
  __bf16* srb  = (__bf16*)(vf + nElem);

  // 1. shift + zigzag + bf16 pack
  prep_xs<<<(int)(nElem / 4 / 256), 256, 0, stream>>>(x, xs);
  // 2. weights -> bf16
  conv_w<<<(int)(4L * kC * kC / 4 / 256), 256, 0, stream>>>(key_w, val_w, rec_w, out_w, wb);
  // 3. k, v, sigmoid(r) via WMMA (B slab LDS-staged, 3 * 96KB dynamic LDS)
  gemm_wmma<3><<<dim3((int)(kBT / 128), kC / 64), 256, 3 * 98304, stream>>>(
      xs, wb, kf, vf, srb);
  // 4. two WKV scans (second one in the re-zigzagged domain, in place)
  wkv_scan<<<(kB * kC) / 256, 256, 0, stream>>>(kf, vf, decay, first, 0, 0);
  wkv_scan<<<(kB * kC) / 256, 256, 0, stream>>>(kf, vf, decay, first, 1, 1);
  // 5. rv = bf16(sr * v)   (reuses xs buffer)
  mul_sr<<<(int)(nElem / 4 / 256), 256, 0, stream>>>(srb, vf, xs);
  // 6. final projection (1 * 96KB dynamic LDS)
  gemm_wmma<1><<<dim3((int)(kBT / 128), kC / 64), 256, 98304, stream>>>(
      xs, wb + 3L * kC * kC, (float*)d_out, nullptr, nullptr);
}
